// SparseConv1x1_33251636806223
// MI455X (gfx1250) — matrix operand order repaired and checked by reading, not compiled, not run on hardware
//
#include <hip/hip_runtime.h>
#include <stdint.h>

// Problem constants (match reference: B,C,H,W = 32,1024,28,28; F = 1024)
#define B_  32
#define C_  1024
#define F_  1024
#define HW_ 784            // 28*28; N per image; 784 = 49*16, 7 tiles of 112

typedef __attribute__((ext_vector_type(16))) __bf16 v16bf;
typedef __attribute__((ext_vector_type(8)))  __bf16 v8bf;
typedef __attribute__((ext_vector_type(8)))  float  v8f;
typedef int v4i __attribute__((vector_size(16)));   // matches builtin param type

// ---- gfx1250 async global->LDS path (guarded; falls back to sync copy) ----
#if defined(__HIP_DEVICE_COMPILE__) && \
    __has_builtin(__builtin_amdgcn_global_load_async_to_lds_b128) && \
    __has_builtin(__builtin_amdgcn_s_wait_asynccnt)
#define USE_ASYNC 1
#else
#define USE_ASYNC 0
#endif

__device__ __forceinline__ void cp16(__bf16* lds_dst, const __bf16* gsrc) {
#if USE_ASYNC
  __builtin_amdgcn_global_load_async_to_lds_b128(
      (__attribute__((address_space(1))) v4i*)gsrc,
      (__attribute__((address_space(3))) v4i*)lds_dst,
      /*imm offset*/0, /*cpol*/0);
#else
  *(uint4*)(void*)lds_dst = *(const uint4*)(const void*)gsrc;
#endif
}

__device__ __forceinline__ void wait_async() {
#if USE_ASYNC
  __builtin_amdgcn_s_wait_asynccnt(0);
#endif
}

// ---- Pass 1a: masked W -> bf16 hi/lo, row-major [F][C] --------------------
__global__ __launch_bounds__(256) void prep_w(const float* __restrict__ k,
                                              const int*   __restrict__ m,
                                              __bf16* __restrict__ hi,
                                              __bf16* __restrict__ lo) {
  int i = blockIdx.x * 256 + threadIdx.x;
  if (i < F_ * C_) {
    float w = m[i] ? k[i] : 0.0f;
    __bf16 h = (__bf16)w;
    hi[i] = h;
    lo[i] = (__bf16)(w - (float)h);
  }
}

// ---- Pass 1b: X [B][C][HW] -> transposed bf16 hi/lo [B][HW][C] ------------
__global__ __launch_bounds__(256) void prep_x(const float* __restrict__ x,
                                              __bf16* __restrict__ hi,
                                              __bf16* __restrict__ lo) {
  __shared__ float tile[32][33];
  const int tx = threadIdx.x & 31, ty = threadIdx.x >> 5;
  const int c0 = blockIdx.x * 32, n0 = blockIdx.y * 32, b = blockIdx.z;
#pragma unroll
  for (int r = 0; r < 4; ++r) {
    int row = r * 8 + ty;                       // c offset within tile
    int n = n0 + tx;
    tile[row][tx] = (n < HW_) ? x[(size_t)b * C_ * HW_ + (size_t)(c0 + row) * HW_ + n]
                              : 0.0f;
  }
  __syncthreads();
#pragma unroll
  for (int r = 0; r < 4; ++r) {
    int row = r * 8 + ty;                       // n offset within tile
    int n = n0 + row;
    if (n < HW_) {
      float w = tile[tx][row];                  // = x[c0+tx][n0+row]
      __bf16 h = (__bf16)w;
      size_t o = (size_t)b * HW_ * C_ + (size_t)n * C_ + (c0 + tx);
      hi[o] = h;
      lo[o] = (__bf16)(w - (float)h);
    }
  }
}

// ---- Pass 2: split-bf16 WMMA GEMM with async double-buffered LDS ----------
// Workgroup = 256 threads (8 waves). Tile: M=128 (f), N=112 (pixels), K-step 32.
// LDS per buffer (bf16 elems): Ahi[128*32]=4096 @0, Alo @4096,
//                              Bhi[112*32]=3584 @8192, Blo @11776  -> 15360
__global__ __launch_bounds__(256) void spmm_wmma(
    const __bf16* __restrict__ w_hi, const __bf16* __restrict__ w_lo,
    const __bf16* __restrict__ x_hi, const __bf16* __restrict__ x_lo,
    const float*  __restrict__ bias, float* __restrict__ out) {
  __shared__ __attribute__((aligned(128))) __bf16 lds[2][15360];

  const int tid  = threadIdx.x;
  const int wv   = tid >> 5;        // wave 0..7 -> M strip
  const int lane = tid & 31;
  const int lrow = lane & 15;
  const int half = lane >> 4;       // K/M half select per ISA VGPR striping
  const int f0 = blockIdx.x * 128;
  const int n0 = blockIdx.y * 112;
  const int b  = blockIdx.z;
  const int xbase = b * HW_ * C_;

  auto stage = [&](int buf, int k0) {
    __bf16* base = &lds[buf][0];
    // A tiles: 128 rows x 32 K of hi+lo = 2 * 512 chunks of 16B
    for (int ch = tid; ch < 512; ch += 256) {
      const int row = ch >> 2;
      const int p   = (ch & 3) * 8;
      const int g   = (f0 + row) * C_ + k0 + p;
      cp16(base + ch * 8,        w_hi + g);
      cp16(base + 4096 + ch * 8, w_lo + g);
    }
    // B tiles: 112 cols x 32 K of hi+lo = 2 * 448 chunks of 16B
    for (int ch = tid; ch < 448; ch += 256) {
      const int row = ch >> 2;
      const int p   = (ch & 3) * 8;
      const int g   = xbase + (n0 + row) * C_ + k0 + p;
      cp16(base + 8192  + ch * 8, x_hi + g);
      cp16(base + 11776 + ch * 8, x_lo + g);
    }
  };

  v8f acc[7] = {};

  stage(0, 0);
  wait_async();
  __syncthreads();

  int cur = 0;
  for (int ks = 0; ks < 32; ++ks) {
    if (ks + 1 < 32) stage(cur ^ 1, (ks + 1) * 32);

    const __bf16* buf  = &lds[cur][0];
    // A fragment (16x32 bf16): lane<16 holds K {0..15}, lane>=16 {16..31}
    const __bf16* arow = buf + (wv * 16 + lrow) * 32;
    union { v16bf v; v8bf h[2]; } aHi, aLo;
    aHi.h[0] = *(const v8bf*)(arow + half * 16);
    aHi.h[1] = *(const v8bf*)(arow + half * 16 + 8);
    const __bf16* arowL = arow + 4096;
    aLo.h[0] = *(const v8bf*)(arowL + half * 16);
    aLo.h[1] = *(const v8bf*)(arowL + half * 16 + 8);

#pragma unroll
    for (int j = 0; j < 7; ++j) {
      // B fragment (32x16 bf16): lane N = lrow, K = half*16 .. half*16+15 contiguous
      const __bf16* bcol = buf + 8192 + (j * 16 + lrow) * 32 + half * 16;
      v16bf bHi = *(const v16bf*)bcol;
      v16bf bLo = *(const v16bf*)(bcol + 3584);
      // split-precision: hi*hi + hi*lo + lo*hi  (~fp32 accuracy)
      acc[j] = __builtin_amdgcn_wmma_f32_16x16x32_bf16(false, aHi.v, false, bHi,
                                                       (short)0, acc[j], false, false);
      acc[j] = __builtin_amdgcn_wmma_f32_16x16x32_bf16(false, aHi.v, false, bLo,
                                                       (short)0, acc[j], false, false);
      acc[j] = __builtin_amdgcn_wmma_f32_16x16x32_bf16(false, aLo.v, false, bHi,
                                                       (short)0, acc[j], false, false);
    }

    wait_async();
    __syncthreads();
    cur ^= 1;
  }

  // Epilogue: C/D layout — lane<16: VGPR i -> M=i, N=lane; lane>=16: M=8+i, N=lane-16
  const int fb = f0 + wv * 16 + half * 8;
#pragma unroll
  for (int j = 0; j < 7; ++j) {
    const int n = n0 + j * 16 + lrow;
    float* op = out + (size_t)b * F_ * HW_ + n;
#pragma unroll
    for (int i = 0; i < 8; ++i) {
      const int f = fb + i;
      float v = acc[j][i] + bias[f];
      op[(size_t)f * HW_] = v > 0.0f ? v : 0.0f;
    }
  }
}

// ---- Host launcher --------------------------------------------------------
// Inputs: d_in[0]=x f32 [32,1024,28,28], d_in[1]=kernel f32 [1024,1024],
//         d_in[2]=mask i32 [1024,1024], d_in[3]=bias f32 [1024]
// Output: f32 [32,1024,28,28]
// Workspace layout (bytes):
//   [0, 2MB)        w_hi  (F*C bf16)
//   [2MB, 4MB)      w_lo
//   [4MB, 4MB+49MB) x_hi  (B*HW*C bf16 = 51,380,224 B)
//   [ ... )         x_lo  (same size)     total ~106.8 MB
extern "C" void kernel_launch(void* const* d_in, const int* in_sizes, int n_in,
                              void* d_out, int out_size, void* d_ws, size_t ws_size,
                              hipStream_t stream) {
  (void)in_sizes; (void)n_in; (void)out_size; (void)ws_size;
  const float* x    = (const float*)d_in[0];
  const float* kern = (const float*)d_in[1];
  const int*   mask = (const int*)d_in[2];
  const float* bias = (const float*)d_in[3];
  float* out = (float*)d_out;

  char* ws = (char*)d_ws;
  __bf16* w_hi = (__bf16*)(ws);
  __bf16* w_lo = w_hi + (size_t)F_ * C_;
  __bf16* x_hi = (__bf16*)(ws + (size_t)4 * 1024 * 1024);
  __bf16* x_lo = x_hi + (size_t)B_ * HW_ * C_;

  prep_w<<<(F_ * C_ + 255) / 256, 256, 0, stream>>>(kern, mask, w_hi, w_lo);

  dim3 gx(C_ / 32, (HW_ + 31) / 32, B_);          // 32 x 25 x 32
  prep_x<<<gx, 256, 0, stream>>>(x, x_hi, x_lo);

  dim3 gg(F_ / 128, HW_ / 112, B_);               // 8 x 7 x 32 = 1792 WGs
  spmm_wmma<<<gg, 256, 0, stream>>>(w_hi, w_lo, x_hi, x_lo, bias, out);
}
